// ParallaxAM_Module_45183055954281
// MI455X (gfx1250) — compile-verified, matrix operand order checked by reading, not run on hardware
//
#include <hip/hip_runtime.h>
#include <hip/hip_bf16.h>

typedef _Float16 v8h  __attribute__((ext_vector_type(8)));
typedef _Float16 v16h __attribute__((ext_vector_type(16)));
typedef float    v8f  __attribute__((ext_vector_type(8)));

#define B_  2
#define C_  256
#define N_  6400
#define CK_ 64
#define CV_ 128
#define CO_ 256

// 0.125 (Ck^-1/2) * log2(e): exp(s/8) == exp2(s * SCALE_LOG2E), single v_exp_f32
#define SCALE_LOG2E 0.1803368801111204f

__device__ __forceinline__ v16h join16(v8h lo, v8h hi) {
  return __builtin_shufflevector(lo, hi, 0,1,2,3,4,5,6,7,8,9,10,11,12,13,14,15);
}
__device__ __forceinline__ v8f zero8() {
  v8f z = {0.f,0.f,0.f,0.f,0.f,0.f,0.f,0.f};
  return z;
}
__device__ __forceinline__ v8f wmma_f16(v16h a, v16h b, v8f c) {
  // D = A(16x32 f16) * B(32x16 f16) + C(16x16 f32)
  return __builtin_amdgcn_wmma_f32_16x16x32_f16(false, a, false, b, (short)0, c, false, false);
}

// Async 16B global -> LDS copy (ASYNCcnt-tracked gfx1250 path).
// VDST = VGPR with LDS byte address, VADDR = 64-bit global address.
__device__ __forceinline__ void async_ld16(void* lds, const void* g) {
  unsigned l = (unsigned)(size_t)lds;    // generic->LDS addr = low 32 bits
  asm volatile("global_load_async_to_lds_b128 %0, %1, off"
               :: "v"(l), "v"(g) : "memory");
}

// ---------------------------------------------------------------------------
// Projection kernel: 64 output channels = W[64,256] * src[:,n] (+BN or +bias)
// mode 0: dst[(b*N+n)*64 + k]  (row-major per position, f16)  -- for Q/K
// mode 1: dst[(b*CV + cvoff + k)*N + n]  (channel-major, f16) -- for V^T
// ---------------------------------------------------------------------------
__global__ void proj64_kernel(const float* __restrict__ src,
                              const float* __restrict__ w,
                              const float* __restrict__ bias,
                              const float* __restrict__ bn_g,
                              const float* __restrict__ bn_b,
                              const float* __restrict__ bn_m,
                              const float* __restrict__ bn_v,
                              _Float16* __restrict__ dst,
                              int mode, int cvoff)
{
  int idx = blockIdx.x * blockDim.x + threadIdx.x;
  if (idx >= B_ * N_) return;
  int b = idx / N_;
  int n = idx % N_;
  const float* sp = src + (size_t)b * C_ * N_ + n;

  float acc[64];
  #pragma unroll
  for (int k = 0; k < 64; ++k) acc[k] = 0.f;

  for (int c = 0; c < C_; ++c) {
    float xv = sp[(size_t)c * N_];          // coalesced across threads (n adjacent)
    const float* wc = w + c;                // uniform -> scalar loads
    #pragma unroll
    for (int k = 0; k < 64; ++k) acc[k] = fmaf(xv, wc[(size_t)k * C_], acc[k]);
  }

  if (mode == 0) {
    _Float16* dp = dst + (size_t)idx * 64;
    #pragma unroll
    for (int k = 0; k < 64; ++k) {
      float s = bn_g[k] * rsqrtf(bn_v[k] + 1e-5f);
      float o = (acc[k] - bn_m[k]) * s + bn_b[k];
      dp[k] = (_Float16)o;
    }
  } else {
    #pragma unroll
    for (int k = 0; k < 64; ++k) {
      float o = acc[k] + bias[k];
      dst[((size_t)b * CV_ + cvoff + k) * N_ + n] = (_Float16)o;
    }
  }
}

// ---------------------------------------------------------------------------
// f32 -> f16 weight conversion (W_w, done once)
// ---------------------------------------------------------------------------
__global__ void cvt_f16_kernel(const float* __restrict__ src,
                               _Float16* __restrict__ dst, int n)
{
  int i = blockIdx.x * blockDim.x + threadIdx.x;
  if (i < n) dst[i] = (_Float16)src[i];
}

// ---------------------------------------------------------------------------
// Flash attention, block-cooperative version.
// Block = 4 waves = 4 consecutive 16-query tiles of ONE batch (100 blocks per
// batch). K/V 32-key tiles are double-buffered in LDS via async global->LDS
// copies (ASYNCcnt) shared by all 4 waves; fragment reads are ds_load_b128
// from bank-conflict-free padded rows (K stride 72 halfs, V stride 40 halfs).
// Softmax without running max (logits O(+-5) after BN; shift-invariant).
// Per key tile per wave: 4 WMMAs for S, 8 WMMAs for P*V, one LDS P round-trip.
// ---------------------------------------------------------------------------
__global__ void __launch_bounds__(128)
flash_kernel(const _Float16* __restrict__ Qm,
             const _Float16* __restrict__ Km,
             const _Float16* __restrict__ Vt,
             _Float16* __restrict__ Ctx)
{
  __shared__ __attribute__((aligned(16))) _Float16 Kl[2][32][72];   // 9.0 KB
  __shared__ __attribute__((aligned(16))) _Float16 Vl[2][128][40];  // 20 KB
  __shared__ __attribute__((aligned(16))) _Float16 Pb[4][16 * 32];  // 4 KB

  const unsigned tid  = threadIdx.x;
  const unsigned lane = tid & 31u;
  const unsigned wib  = tid >> 5;
  const int b  = (int)(blockIdx.x / (N_ / 16 / 4));                 // /100
  const int qb = ((int)(blockIdx.x % (N_ / 16 / 4)) * 4 + (int)wib) * 16;
  const unsigned l16 = lane & 15u;
  const unsigned h   = lane >> 4;       // 0 or 1

  const _Float16* Kb = Km + (size_t)b * N_ * CK_;
  const _Float16* Vb = Vt + (size_t)b * CV_ * N_;

  // Q A-fragments (16x32 f16, two Ck chunks), contiguous 16B loads per lane
  const _Float16* qrow = Qm + ((size_t)b * N_ + qb + l16) * CK_;
  v16h aq0 = join16(*(const v8h*)(qrow +       h * 8), *(const v8h*)(qrow + 16 + h * 8));
  v16h aq1 = join16(*(const v8h*)(qrow + 32 +  h * 8), *(const v8h*)(qrow + 48 + h * 8));

  v8f o[8];
  #pragma unroll
  for (int j = 0; j < 8; ++j) o[j] = zero8();
  float psum[8];
  #pragma unroll
  for (int r = 0; r < 8; ++r) psum[r] = 0.f;

  // issue async staging of one 32-key tile: 6 ASYNC ops per thread per tile
  auto stage = [&](int buf, int kb) {
    // K tile: 32 rows x 64 halfs = 256 x 16B segments, 2 per thread
    #pragma unroll
    for (int s = 0; s < 2; ++s) {
      unsigned seg = tid * 2 + s;              // 0..255
      unsigned r = seg >> 3, c = seg & 7;      // key row, 8-half column group
      async_ld16(&Kl[buf][r][c * 8], Kb + (size_t)(kb + (int)r) * CK_ + c * 8);
    }
    // V tile: thread tid owns cv row tid: 32 halfs = 4 x 16B segments
    #pragma unroll
    for (int s = 0; s < 4; ++s) {
      async_ld16(&Vl[buf][tid][s * 8], Vb + (size_t)tid * N_ + kb + s * 8);
    }
  };

  stage(0, 0);
  int cur = 0;

  for (int kb = 0; kb < N_; kb += 32) {
    if (kb + 32 < N_) {
      stage(cur ^ 1, kb + 32);                       // prefetch next tile
      asm volatile("s_wait_asynccnt 6" ::: "memory"); // in-order: cur resident
    } else {
      asm volatile("s_wait_asynccnt 0" ::: "memory");
    }
    __syncthreads();   // staged tile visible to all 4 waves

    // ---- load ALL four K B-fragments first (lets ds_loads pipeline), ----
    // ---- then run the 4 S-WMMAs                                       ----
    const _Float16* kr0 = &Kl[cur][l16][0];
    const _Float16* kr1 = &Kl[cur][16 + l16][0];
    v16h bk00 = join16(*(const v8h*)(kr0 +      h * 8), *(const v8h*)(kr0 + 16 + h * 8));
    v16h bk01 = join16(*(const v8h*)(kr0 + 32 + h * 8), *(const v8h*)(kr0 + 48 + h * 8));
    v16h bk10 = join16(*(const v8h*)(kr1 +      h * 8), *(const v8h*)(kr1 + 16 + h * 8));
    v16h bk11 = join16(*(const v8h*)(kr1 + 32 + h * 8), *(const v8h*)(kr1 + 48 + h * 8));

    v8f c0, c1;
    {
      v8f z = zero8();
      c0 = wmma_f16(aq0, bk00, z);
      c1 = wmma_f16(aq0, bk10, z);
      c0 = wmma_f16(aq1, bk01, c0);
      c1 = wmma_f16(aq1, bk11, c1);
    }

    // ---- p = exp(s/8) = exp2(s * SCALE_LOG2E); accumulate row sums ----
    #pragma unroll
    for (int r = 0; r < 8; ++r) {
      float p0 = __builtin_amdgcn_exp2f(c0[r] * SCALE_LOG2E);
      float p1 = __builtin_amdgcn_exp2f(c1[r] * SCALE_LOG2E);
      psum[r] += p0 + p1;
      int row = r + 8 * (int)h;
      Pb[wib][row * 32 +      l16] = (_Float16)p0;
      Pb[wib][row * 32 + 16 + l16] = (_Float16)p1;
    }

    asm volatile("s_wait_dscnt 0" ::: "memory");

    // reload P as a 16x32 A-fragment (contiguous 16B LDS reads per lane)
    const _Float16* pr = &Pb[wib][l16 * 32 + h * 8];
    v16h pA = join16(*(const v8h*)pr, *(const v8h*)(pr + 16));

    // ---- O += P * V ; 8 column groups over Cv=128, loads one step ahead ----
    const _Float16* vr0 = &Vl[cur][l16][0];
    v16h bv = join16(*(const v8h*)(vr0 + h * 8), *(const v8h*)(vr0 + 16 + h * 8));
    #pragma unroll
    for (int j = 0; j < 8; ++j) {
      v16h bv_cur = bv;
      if (j < 7) {
        const _Float16* vr = &Vl[cur][(j + 1) * 16 + l16][0];
        bv = join16(*(const v8h*)(vr + h * 8), *(const v8h*)(vr + 16 + h * 8));
      }
      o[j] = wmma_f16(pA, bv_cur, o[j]);
    }

    __syncthreads();   // all waves done reading buf[cur] before it is restaged
    cur ^= 1;
  }

  // ---- one lane-reduction per row at the very end ----
  float rinv[8];
  #pragma unroll
  for (int r = 0; r < 8; ++r) {
    float t = psum[r];
    #pragma unroll
    for (int m = 8; m >= 1; m >>= 1) t += __shfl_xor(t, m, 32);
    rinv[r] = 1.0f / t;
  }

  // ---- epilogue: normalize and store ctx[(b, n), cv] as f16 ----
  #pragma unroll
  for (int j = 0; j < 8; ++j) {
    #pragma unroll
    for (int r = 0; r < 8; ++r) {
      int m = r + 8 * (int)h;
      Ctx[((size_t)b * N_ + qb + m) * CV_ + j * 16 + l16] = (_Float16)(o[j][r] * rinv[r]);
    }
  }
}

// ---------------------------------------------------------------------------
// Output projection: out = gamma * (Wh[256,128] @ ctx + W_b) + x
// One wave = 16 output channels x 16 positions, K = Cv = 128 (4 WMMA chunks).
// Wh is the pre-converted f16 copy of W_w.
// ---------------------------------------------------------------------------
__global__ void __launch_bounds__(128)
outproj_kernel(const _Float16* __restrict__ Wh,
               const float* __restrict__ Wb,
               const _Float16* __restrict__ Ctx,
               const float* __restrict__ x,
               const float* __restrict__ gamma,
               float* __restrict__ out)
{
  const unsigned lane = threadIdx.x & 31u;
  const unsigned wg   = blockIdx.x * (blockDim.x >> 5) + (threadIdx.x >> 5);
  const int tilesN = N_ / 16;                       // 400
  const int b   = (int)(wg / ((CO_ / 16) * tilesN));
  int rem       = (int)(wg % ((CO_ / 16) * tilesN));
  const int cob = (rem / tilesN) * 16;
  const int pb  = (rem % tilesN) * 16;
  const unsigned l16 = lane & 15u;
  const unsigned h   = lane >> 4;

  v8f acc = zero8();
  const _Float16* wr = Wh  + (size_t)(cob + l16) * CV_;
  const _Float16* cr = Ctx + ((size_t)b * N_ + pb + l16) * CV_;

  #pragma unroll
  for (int cc = 0; cc < 4; ++cc) {
    const _Float16* wp = wr + cc * 32 + h * 8;
    v16h a  = join16(*(const v8h*)wp, *(const v8h*)(wp + 16));
    const _Float16* cp = cr + cc * 32 + h * 8;
    v16h bb = join16(*(const v8h*)cp, *(const v8h*)(cp + 16));
    acc = wmma_f16(a, bb, acc);
  }

  float gm = gamma[0];
  #pragma unroll
  for (int r = 0; r < 8; ++r) {
    int co = cob + r + 8 * (int)h;
    size_t oi = ((size_t)b * CO_ + co) * N_ + pb + l16;
    out[oi] = gm * (acc[r] + Wb[co]) + x[oi];
  }
}

// ---------------------------------------------------------------------------
extern "C" void kernel_launch(void* const* d_in, const int* in_sizes, int n_in,
                              void* d_out, int out_size, void* d_ws, size_t ws_size,
                              hipStream_t stream)
{
  const float* x     = (const float*)d_in[0];
  const float* y     = (const float*)d_in[1];
  const float* fk_w  = (const float*)d_in[2];
  const float* fk_g  = (const float*)d_in[3];
  const float* fk_b  = (const float*)d_in[4];
  const float* fk_m  = (const float*)d_in[5];
  const float* fk_v  = (const float*)d_in[6];
  const float* fq_w  = (const float*)d_in[7];
  const float* fq_g  = (const float*)d_in[8];
  const float* fq_b  = (const float*)d_in[9];
  const float* fq_m  = (const float*)d_in[10];
  const float* fq_v  = (const float*)d_in[11];
  const float* fv_w  = (const float*)d_in[12];
  const float* fv_b  = (const float*)d_in[13];
  const float* W_w   = (const float*)d_in[14];
  const float* W_b   = (const float*)d_in[15];
  const float* gamma = (const float*)d_in[16];
  float* out = (float*)d_out;

  char* ws = (char*)d_ws;
  size_t off = 0;
  auto alloc = [&](size_t bytes) -> void* {
    void* p = ws + off;
    off = (off + bytes + 255) & ~(size_t)255;
    return p;
  };
  _Float16* Qm  = (_Float16*)alloc((size_t)B_ * N_ * CK_ * sizeof(_Float16));
  _Float16* Km  = (_Float16*)alloc((size_t)B_ * N_ * CK_ * sizeof(_Float16));
  _Float16* Vt  = (_Float16*)alloc((size_t)B_ * CV_ * N_ * sizeof(_Float16));
  _Float16* Ctx = (_Float16*)alloc((size_t)B_ * N_ * CV_ * sizeof(_Float16));
  _Float16* Wh  = (_Float16*)alloc((size_t)CO_ * CV_ * sizeof(_Float16));
  (void)in_sizes; (void)n_in; (void)out_size; (void)ws_size;

  // projections
  dim3 pblk(256), pgrd((B_ * N_ + 255) / 256);
  proj64_kernel<<<pgrd, pblk, 0, stream>>>(y, fq_w, nullptr, fq_g, fq_b, fq_m, fq_v, Qm, 0, 0);
  proj64_kernel<<<pgrd, pblk, 0, stream>>>(x, fk_w, nullptr, fk_g, fk_b, fk_m, fk_v, Km, 0, 0);
  proj64_kernel<<<pgrd, pblk, 0, stream>>>(y, fv_w,            fv_b,      nullptr, nullptr, nullptr, nullptr, Vt, 1, 0);
  proj64_kernel<<<pgrd, pblk, 0, stream>>>(y, fv_w + 64 * C_,  fv_b + 64, nullptr, nullptr, nullptr, nullptr, Vt, 1, 64);

  // one-time W_w -> f16
  cvt_f16_kernel<<<dim3((CO_ * CV_ + 255) / 256), dim3(256), 0, stream>>>(W_w, Wh, CO_ * CV_);

  // flash attention: 100 blocks per batch, 4 waves (= 4 query tiles) per block
  flash_kernel<<<dim3(B_ * (N_ / 16 / 4)), dim3(128), 0, stream>>>(Qm, Km, Vt, Ctx);

  // output projection + residual: B * 16 * 400 wave tiles, 4 waves per block
  outproj_kernel<<<dim3((B_ * (CO_ / 16) * (N_ / 16)) / 4), dim3(128), 0, stream>>>(
      Wh, W_b, Ctx, x, gamma, out);
}
